// Block_40200893891060
// MI455X (gfx1250) — matrix operand order repaired and checked by reading, not compiled, not run on hardware
//
#include <hip/hip_runtime.h>
#include <hip/hip_bf16.h>

// ---------------------------------------------------------------------------
// ConvNeXt block for MI455X (gfx1250, wave32).
// dwconv7x7 -> LayerNorm -> GEMM1(bf16 WMMA)+GELU -> GEMM2(bf16 WMMA)*gamma
// -> async-LDS transpose + residual.
// GEMMs: v_wmma_f32_16x16x32_bf16, 32Mx64N wave tile, double-buffered K loop.
// ---------------------------------------------------------------------------

typedef __attribute__((ext_vector_type(16))) __bf16 v16bf;
typedef __attribute__((ext_vector_type(8)))  __bf16 v8bf;
typedef __attribute__((ext_vector_type(8)))  float  v8f;

#define DIM   384
#define HID   1536
#define IMG   56
#define HW    3136          // 56*56
#define NB    16
#define MTOK  50176         // NB*HW

// ---- WMMA fragment loaders (contiguous-K order, same for A and B) ------------------
// A (16x32, MxK): lane l<16 holds row M=l, K=0..15;
//                 lane l>=16 holds row M=l-16, K=16..31 (matches B).
__device__ __forceinline__ v16bf load_a_frag(const __bf16* __restrict__ Arow,
                                             int k, int half) {
  return *(const v16bf*)(Arow + k + half * 16);
}
// B (32x16, KxN): lane l holds column N=l%16; lanes 0-15 hold K=0..15,
// lanes 16-31 hold K=16..31 (contiguous in K -> single 32B load from B^T row).
__device__ __forceinline__ v16bf load_b_frag(const __bf16* __restrict__ BTrow,
                                             int k, int half) {
  return *(const v16bf*)(BTrow + k + half * 16);
}

// ---- 1) weight prep: transpose + f32->bf16 ---------------------------------
__global__ void __launch_bounds__(256)
prep_weights(const float* __restrict__ w1, const float* __restrict__ w2,
             __bf16* __restrict__ w1t, __bf16* __restrict__ w2t) {
  int i = blockIdx.x * 256 + threadIdx.x;
  if (i < DIM * HID) {
    int f = i / DIM, c = i % DIM;          // w1t[f][c] = w1[c][f]
    w1t[i] = (__bf16)w1[c * HID + f];
    int c2 = i / HID, f2 = i % HID;        // w2t[c][f] = w2[f][c]
    w2t[i] = (__bf16)w2[f2 * DIM + c2];
  }
}

// ---- 2) depthwise 7x7 conv + bias, NCHW f32 --------------------------------
__global__ void __launch_bounds__(64)
dwconv(const float* __restrict__ x, const float* __restrict__ dw_w,
       const float* __restrict__ dw_b, float* __restrict__ conv) {
  const int b = blockIdx.x;                 // over NB*DIM*IMG
  const int h = b % IMG;
  const int c = (b / IMG) % DIM;
  const int n = b / (IMG * DIM);
  __shared__ float xr[7 * 62];              // rows h-3..h+3, w -3..58
  __shared__ float wgt[49];
  const int tid = threadIdx.x;
  const float* xc = x + ((size_t)(n * DIM + c)) * HW;
  for (int idx = tid; idx < 7 * 62 + 49; idx += 64) {
    if (idx < 7 * 62) {
      int dy = idx / 62, j = idx % 62;
      int hy = h + dy - 3, wx = j - 3;
      float v = 0.0f;
      if (hy >= 0 && hy < IMG && wx >= 0 && wx < IMG) v = xc[hy * IMG + wx];
      xr[idx] = v;
    } else {
      int wi = idx - 7 * 62;
      wgt[wi] = dw_w[wi * DIM + c];
    }
  }
  __syncthreads();
  if (tid < IMG) {
    float s = dw_b[c];
    #pragma unroll
    for (int dy = 0; dy < 7; ++dy)
      #pragma unroll
      for (int dx = 0; dx < 7; ++dx)
        s += xr[dy * 62 + tid + dx] * wgt[dy * 7 + dx];
    conv[((size_t)(n * DIM + c) * IMG + h) * IMG + tid] = s;
  }
}

// ---- 3) LayerNorm over channels, NCHW f32 -> NHWC bf16 ---------------------
__global__ void __launch_bounds__(448)
ln_kernel(const float* __restrict__ conv, const float* __restrict__ ln_w,
          const float* __restrict__ ln_b, __bf16* __restrict__ A) {
  const int b = blockIdx.x;                 // over NB*IMG*2 half rows
  const int hr = b & 1;
  const int h = (b >> 1) % IMG;
  const int n = b / (2 * IMG);
  const int w0 = hr * 28;
  __shared__ float tile[DIM * 28];
  __shared__ float mu[28], rs[28];
  const int tid = threadIdx.x;
  const int p = tid % 28, cg = tid / 28;    // 16 channel groups
  for (int c = cg; c < DIM; c += 16)
    tile[c * 28 + p] = conv[(((size_t)n * DIM + c) * IMG + h) * IMG + w0 + p];
  __syncthreads();
  if (tid < 28) {
    float s = 0.f, sq = 0.f;
    for (int c = 0; c < DIM; ++c) {
      float v = tile[c * 28 + tid];
      s += v; sq += v * v;
    }
    float m = s * (1.0f / DIM);
    float var = sq * (1.0f / DIM) - m * m;
    mu[tid] = m;
    rs[tid] = rsqrtf(var + 1e-6f);
  }
  __syncthreads();
  const size_t tok0 = ((size_t)n * IMG + h) * IMG + w0;
  for (int idx = tid; idx < DIM * 28; idx += 448) {
    int c = idx % DIM, pp = idx / DIM;
    float v = (tile[c * 28 + pp] - mu[pp]) * rs[pp] * ln_w[c] + ln_b[c];
    A[(tok0 + pp) * DIM + c] = (__bf16)v;
  }
}

// ---- 4) GEMM1: H[M,1536] = gelu(A[M,384] @ W1 + b1), bf16 WMMA -------------
// 8 waves: 4 along M x 2 along N; wave tile 32Mx64N; block tile 128x128.
__global__ void __launch_bounds__(256)
gemm1_gelu(const __bf16* __restrict__ A, const __bf16* __restrict__ W1T,
           const float* __restrict__ bias1, __bf16* __restrict__ Hout) {
  const int lane = threadIdx.x & 31;
  const int wid  = threadIdx.x >> 5;
  const int half = lane >> 4;
  const int lr   = lane & 15;
  const int m0 = blockIdx.y * 128 + (wid & 3) * 32;
  const int n0 = blockIdx.x * 128 + (wid >> 2) * 64;
  const __bf16* Ar0 = A + (size_t)(m0 + lr) * DIM;
  const __bf16* Ar1 = A + (size_t)(m0 + 16 + lr) * DIM;
  const __bf16* Br[4] = {
    W1T + (size_t)(n0 +  0 + lr) * DIM, W1T + (size_t)(n0 + 16 + lr) * DIM,
    W1T + (size_t)(n0 + 32 + lr) * DIM, W1T + (size_t)(n0 + 48 + lr) * DIM };
  v8f acc[2][4] = {};
  v16bf a0 = load_a_frag(Ar0, 0, half);
  v16bf a1 = load_a_frag(Ar1, 0, half);
  v16bf bfr[4];
  #pragma unroll
  for (int t = 0; t < 4; ++t) bfr[t] = load_b_frag(Br[t], 0, half);
  #pragma unroll 2
  for (int k = 32; k < DIM; k += 32) {
    v16bf na0 = load_a_frag(Ar0, k, half);
    v16bf na1 = load_a_frag(Ar1, k, half);
    v16bf nb[4];
    #pragma unroll
    for (int t = 0; t < 4; ++t) nb[t] = load_b_frag(Br[t], k, half);
    #pragma unroll
    for (int t = 0; t < 4; ++t) {
      acc[0][t] = __builtin_amdgcn_wmma_f32_16x16x32_bf16(
          false, a0, false, bfr[t], (short)0, acc[0][t], false, false);
      acc[1][t] = __builtin_amdgcn_wmma_f32_16x16x32_bf16(
          false, a1, false, bfr[t], (short)0, acc[1][t], false, false);
    }
    a0 = na0; a1 = na1;
    #pragma unroll
    for (int t = 0; t < 4; ++t) bfr[t] = nb[t];
  }
  #pragma unroll
  for (int t = 0; t < 4; ++t) {
    acc[0][t] = __builtin_amdgcn_wmma_f32_16x16x32_bf16(
        false, a0, false, bfr[t], (short)0, acc[0][t], false, false);
    acc[1][t] = __builtin_amdgcn_wmma_f32_16x16x32_bf16(
        false, a1, false, bfr[t], (short)0, acc[1][t], false, false);
  }
  #pragma unroll
  for (int i = 0; i < 2; ++i) {
    const int mb = m0 + i * 16 + half * 8;
    #pragma unroll
    for (int t = 0; t < 4; ++t) {
      const int nn = n0 + t * 16 + lr;
      const float bias = bias1[nn];
      #pragma unroll
      for (int v = 0; v < 8; ++v) {
        float xv = acc[i][t][v] + bias;
        float g = 0.5f * xv * (1.0f + erff(xv * 0.70710678118654752f));
        Hout[(size_t)(mb + v) * HID + nn] = (__bf16)g;
      }
    }
  }
}

// ---- 5) GEMM2: Y[M,384] = (H[M,1536] @ W2 + b2) * gamma, bf16 WMMA ---------
__global__ void __launch_bounds__(256)
gemm2_scale(const __bf16* __restrict__ Hin, const __bf16* __restrict__ W2T,
            const float* __restrict__ bias2, const float* __restrict__ gamma,
            float* __restrict__ Y) {
  const int lane = threadIdx.x & 31;
  const int wid  = threadIdx.x >> 5;
  const int half = lane >> 4;
  const int lr   = lane & 15;
  const int m0 = blockIdx.y * 128 + (wid & 3) * 32;
  const int n0 = blockIdx.x * 128 + (wid >> 2) * 64;
  const __bf16* Ar0 = Hin + (size_t)(m0 + lr) * HID;
  const __bf16* Ar1 = Hin + (size_t)(m0 + 16 + lr) * HID;
  const __bf16* Br[4] = {
    W2T + (size_t)(n0 +  0 + lr) * HID, W2T + (size_t)(n0 + 16 + lr) * HID,
    W2T + (size_t)(n0 + 32 + lr) * HID, W2T + (size_t)(n0 + 48 + lr) * HID };
  v8f acc[2][4] = {};
  v16bf a0 = load_a_frag(Ar0, 0, half);
  v16bf a1 = load_a_frag(Ar1, 0, half);
  v16bf bfr[4];
  #pragma unroll
  for (int t = 0; t < 4; ++t) bfr[t] = load_b_frag(Br[t], 0, half);
  #pragma unroll 2
  for (int k = 32; k < HID; k += 32) {
    v16bf na0 = load_a_frag(Ar0, k, half);
    v16bf na1 = load_a_frag(Ar1, k, half);
    v16bf nb[4];
    #pragma unroll
    for (int t = 0; t < 4; ++t) nb[t] = load_b_frag(Br[t], k, half);
    #pragma unroll
    for (int t = 0; t < 4; ++t) {
      acc[0][t] = __builtin_amdgcn_wmma_f32_16x16x32_bf16(
          false, a0, false, bfr[t], (short)0, acc[0][t], false, false);
      acc[1][t] = __builtin_amdgcn_wmma_f32_16x16x32_bf16(
          false, a1, false, bfr[t], (short)0, acc[1][t], false, false);
    }
    a0 = na0; a1 = na1;
    #pragma unroll
    for (int t = 0; t < 4; ++t) bfr[t] = nb[t];
  }
  #pragma unroll
  for (int t = 0; t < 4; ++t) {
    acc[0][t] = __builtin_amdgcn_wmma_f32_16x16x32_bf16(
        false, a0, false, bfr[t], (short)0, acc[0][t], false, false);
    acc[1][t] = __builtin_amdgcn_wmma_f32_16x16x32_bf16(
        false, a1, false, bfr[t], (short)0, acc[1][t], false, false);
  }
  #pragma unroll
  for (int i = 0; i < 2; ++i) {
    const int mb = m0 + i * 16 + half * 8;
    #pragma unroll
    for (int t = 0; t < 4; ++t) {
      const int nn = n0 + t * 16 + lr;
      const float bias = bias2[nn], g = gamma[nn];
      #pragma unroll
      for (int v = 0; v < 8; ++v)
        Y[(size_t)(mb + v) * DIM + nn] = (acc[i][t][v] + bias) * g;
    }
  }
}

// ---- 6) NHWC -> NCHW transpose + residual, async-LDS staged ----------------
__global__ void __launch_bounds__(256)
transpose_residual(const float* __restrict__ Y, const float* __restrict__ x,
                   float* __restrict__ out) {
  __shared__ __align__(16) float tile[32 * 36];   // 36-float row stride:
  const int n  = blockIdx.z;                      // 16B-aligned rows, 2-way
  const int c0 = blockIdx.y * 32;                 // conflicts on column reads
  const int p0 = blockIdx.x * 32;
  const int tid = threadIdx.x;
  // async load: 256 lanes x 16B = 32x32 f32 tile straight into LDS
  const int pl = tid >> 3;          // token row 0..31
  const int c4 = (tid & 7) * 4;     // channel group 0,4,...,28
  const float* gptr = Y + ((size_t)n * HW + p0 + pl) * DIM + c0 + c4;
  unsigned lds_off = (unsigned)(uintptr_t)(&tile[pl * 36 + c4]);
  asm volatile("global_load_async_to_lds_b128 %0, %1, off"
               :: "v"(lds_off), "v"(gptr) : "memory");
  asm volatile("s_wait_asynccnt 0" ::: "memory");
  __syncthreads();
  const int tx = tid & 31, ty = tid >> 5;
  #pragma unroll
  for (int i = 0; i < 4; ++i) {
    int c = c0 + ty + i * 8;
    size_t o = ((size_t)n * DIM + c) * HW + p0 + tx;
    out[o] = tile[tx * 36 + ty + i * 8] + x[o];
  }
}

// ---------------------------------------------------------------------------
extern "C" void kernel_launch(void* const* d_in, const int* in_sizes, int n_in,
                              void* d_out, int out_size, void* d_ws, size_t ws_size,
                              hipStream_t stream) {
  const float* x     = (const float*)d_in[0];
  const float* dw_w  = (const float*)d_in[1];
  const float* dw_b  = (const float*)d_in[2];
  const float* ln_w  = (const float*)d_in[3];
  const float* ln_b  = (const float*)d_in[4];
  const float* w1    = (const float*)d_in[5];
  const float* b1    = (const float*)d_in[6];
  const float* w2    = (const float*)d_in[7];
  const float* b2    = (const float*)d_in[8];
  const float* gamma = (const float*)d_in[9];
  float* out = (float*)d_out;

  // workspace layout (all offsets 256B aligned)
  char* ws = (char*)d_ws;
  __bf16* W1T   = (__bf16*)(ws);                         //  1,179,648 B
  __bf16* W2T   = (__bf16*)(ws + 1179648);               //  1,179,648 B
  __bf16* Abuf  = (__bf16*)(ws + 2359296);               // 38,535,168 B
  __bf16* Hbuf  = (__bf16*)(ws + 40894464);              // 154,140,672 B
  float*  CONVY = (float*) (ws + 195035136);             // 77,070,336 B (conv, later Y)

  prep_weights<<<(DIM * HID + 255) / 256, 256, 0, stream>>>(w1, w2, W1T, W2T);
  dwconv<<<NB * DIM * IMG, 64, 0, stream>>>(x, dw_w, dw_b, CONVY);
  ln_kernel<<<NB * IMG * 2, 448, 0, stream>>>(CONVY, ln_w, ln_b, Abuf);
  gemm1_gelu<<<dim3(HID / 128, MTOK / 128), 256, 0, stream>>>(Abuf, W1T, b1, Hbuf);
  gemm2_scale<<<dim3(DIM / 128, MTOK / 128), 256, 0, stream>>>(Hbuf, W2T, b2, gamma, CONVY);
  transpose_residual<<<dim3(HW / 32, DIM / 32, NB), 256, 0, stream>>>(CONVY, x, out);
}
